// MixRecModel_24867860644153
// MI455X (gfx1250) — matrix operand order repaired and checked by reading, not compiled, not run on hardware
//
#include <hip/hip_runtime.h>
#include <hip/hip_bf16.h>
#include <math.h>

// ---------------- constants from the reference ----------------
#define NUM_USERS   100000
#define NUM_ITEMS   50000
#define DIM         64
#define N_LAYERS    3
#define BATCH       4096
#define N_NODES     (NUM_USERS + NUM_ITEMS + 1)     // 150001
#define SSL_LAMBDA  0.1f
#define INV_TEMP    5.0f                            // 1/0.2
#define REG_C       0.0001f
#define ALPHA_GCN   0.25f                           // 1/(N_LAYERS+1)

typedef __attribute__((ext_vector_type(16))) _Float16 v16h;
typedef __attribute__((ext_vector_type(8)))  float    v8f;
typedef __attribute__((ext_vector_type(4)))  float    f32x4;
typedef __attribute__((ext_vector_type(2)))  float    f32x2;

// ---------------- workspace layout (float offsets) ----------------
static const size_t NN64      = (size_t)N_NODES * DIM;   // 9,600,064 (256B aligned)
static const size_t OFF_ACC   = 0;
static const size_t OFF_CUR   = NN64;
static const size_t OFF_NXT   = 2 * NN64;
static const size_t OFF_S     = OFF_CUR;         // aliases cur+nxt (dead after GCN)
static const size_t BROW      = (size_t)BATCH * DIM;  // 262,144
static const size_t OFF_UE    = 3 * NN64;
static const size_t OFF_PE    = OFF_UE   + BROW;
static const size_t OFF_NE    = OFF_PE   + BROW;
static const size_t OFF_NUEF  = OFF_NE   + BROW;
static const size_t OFF_NMUF  = OFF_NUEF + BROW;
static const size_t OFF_NPEF  = OFF_NMUF + BROW;
static const size_t OFF_NMPF  = OFF_NPEF + BROW;
static const size_t OFF_H16   = OFF_NMPF + BROW;      // 4 half-buffers = 2*BROW floats
static const size_t OFF_NCM   = OFF_H16  + 2 * BROW;  // 128 floats (u then p)
static const size_t OFF_SIMS  = OFF_NCM  + 128;       // 8 arrays x 4096
static const size_t OFF_ROWS  = OFF_SIMS + 8 * BATCH; // rowMU,rowEU,rowMP,rowEP
static const size_t OFF_ACCS  = OFF_ROWS + 4 * BATCH; // 64 accumulator floats
// accums: [0]=sum(beta_i) [1]=reg_sq [2]=sum bpr_pos [3]=sum bpr_neg [4]=cl_u [5]=cl_p

// ---------------- device helpers ----------------
__device__ __forceinline__ float neg_log_sigmoid(float z) {
    return fmaxf(-z, 0.0f) + log1pf(expf(-fabsf(z)));
}
__device__ __forceinline__ float lse3(float M, float E, float x, float y) {
    float m2 = fmaxf(M, fmaxf(x, y));
    return m2 + logf(E * expf(M - m2) + expf(x - m2) + expf(y - m2));
}

// CDNA5 async global->LDS copy (16B per enabled lane), tracked with ASYNCcnt.
// VDST = 32-bit LDS byte address (low 32 bits of the generic LDS pointer),
// VADDR = 64-bit global address.  (CDNA5 ISA)
__device__ __forceinline__ void async_b128_to_lds(const void* gsrc, void* lds_dst) {
    unsigned lds_addr = (unsigned)(size_t)lds_dst;
    unsigned long long gaddr = (unsigned long long)(size_t)gsrc;
    asm volatile("global_load_async_to_lds_b128 %0, %1, off"
                 :: "v"(lds_addr), "v"(gaddr) : "memory");
}
__device__ __forceinline__ void wait_asynccnt0() {
#if __has_builtin(__builtin_amdgcn_s_wait_asynccnt)
    __builtin_amdgcn_s_wait_asynccnt(0);
#else
    asm volatile("s_wait_asynccnt 0x0" ::: "memory");
#endif
}

__device__ __forceinline__ v8f wmma_f16(const v16h& a, const v16h& b, const v8f& c) {
    return __builtin_amdgcn_wmma_f32_16x16x32_f16(false, a, false, b, (short)0, c,
                                                  false, false);
}

// ---------------- kernels ----------------
__global__ void k_zero(float* __restrict__ p, size_t n) {
    size_t i = (size_t)blockIdx.x * blockDim.x + threadIdx.x;
    if (i < n) p[i] = 0.0f;
}

__global__ void k_init(const float* __restrict__ user_emb, const float* __restrict__ item_emb,
                       float* __restrict__ acc, float* __restrict__ cur) {
    size_t i = (size_t)blockIdx.x * blockDim.x + threadIdx.x;
    if (i >= NN64) return;
    float x = (i < (size_t)NUM_USERS * DIM) ? user_emb[i] : item_emb[i - (size_t)NUM_USERS * DIM];
    cur[i] = x;
    acc[i] = ALPHA_GCN * x;
}

// one wave per edge; each lane handles 2 dims. L2-resident accumulator (38MB < 192MB L2).
__global__ void k_scatter(const int* __restrict__ rows, const int* __restrict__ cols,
                          const float* __restrict__ vals,
                          const float* __restrict__ cur, float* __restrict__ nxt, int nnz) {
    int t = blockIdx.x * blockDim.x + threadIdx.x;
    int e = t >> 5;
    if (e >= nnz) return;
    int lane = t & 31;
    int r = rows[e], c = cols[e];
    float v = vals[e];
    f32x2 x = *(const f32x2*)&cur[(size_t)c * DIM + lane * 2];
    atomicAdd(&nxt[(size_t)r * DIM + lane * 2 + 0], v * x.x);
    atomicAdd(&nxt[(size_t)r * DIM + lane * 2 + 1], v * x.y);
}

__global__ void k_update(float* __restrict__ acc, float* __restrict__ cur,
                         const float* __restrict__ nxt) {
    size_t i = (size_t)blockIdx.x * blockDim.x + threadIdx.x;
    if (i >= NN64) return;
    float v = nxt[i];
    acc[i] += ALPHA_GCN * v;
    cur[i] = v;
}

__global__ __launch_bounds__(64) void k_gather(const float* __restrict__ acc,
        const int* __restrict__ user, const int* __restrict__ pos, const int* __restrict__ neg,
        const float* __restrict__ user_emb, const float* __restrict__ item_emb,
        float* __restrict__ ue, float* __restrict__ pe, float* __restrict__ ne,
        float* __restrict__ accums) {
    int b = blockIdx.x, d = threadIdx.x;
    int uu = user[b], ii = pos[b], nn = neg[b];
    ue[b * DIM + d] = acc[(size_t)uu * DIM + d];
    pe[b * DIM + d] = acc[((size_t)NUM_USERS + ii) * DIM + d];
    ne[b * DIM + d] = acc[((size_t)NUM_USERS + nn) * DIM + d];
    float a1 = user_emb[(size_t)uu * DIM + d];
    float a2 = item_emb[(size_t)ii * DIM + d];
    float a3 = item_emb[(size_t)nn * DIM + d];
    __shared__ float red[64];
    red[d] = a1 * a1 + a2 * a2 + a3 * a3;
    __syncthreads();
    for (int s = 32; s > 0; s >>= 1) { if (d < s) red[d] += red[d + s]; __syncthreads(); }
    if (d == 0) atomicAdd(&accums[1], red[0]);
}

// per-row: mixups, BPR scores, L2 norms, normalized f32 + f16 copies
__global__ __launch_bounds__(64) void k_batch(const float* __restrict__ ue,
        const float* __restrict__ pe, const float* __restrict__ ne,
        const int* __restrict__ perm, const float* __restrict__ beta_u,
        const float* __restrict__ beta_i,
        float* __restrict__ nueF, float* __restrict__ nmuF,
        float* __restrict__ npeF, float* __restrict__ nmpF,
        _Float16* __restrict__ nueH, _Float16* __restrict__ nmuH,
        _Float16* __restrict__ npeH, _Float16* __restrict__ nmpH,
        float* __restrict__ accums) {
    int b = blockIdx.x, d = threadIdx.x;
    int pb = perm[b];
    float u  = ue[b * DIM + d],  p  = pe[b * DIM + d],  n  = ne[b * DIM + d];
    float up = ue[pb * DIM + d], pp = pe[pb * DIM + d], np_ = ne[pb * DIM + d];
    float bu = beta_u[b], bi = beta_i[b];
    float umix = bu * u + (1.0f - bu) * up;
    float pmix = bi * p + (1.0f - bi) * pp;
    float nmix = bi * n + (1.0f - bi) * np_;
    __shared__ float red[7][64];
    red[0][d] = u * p;    red[1][d] = u * n;       red[2][d] = u * nmix;
    red[3][d] = u * u;    red[4][d] = umix * umix; red[5][d] = p * p;
    red[6][d] = pmix * pmix;
    __syncthreads();
    for (int s = 32; s > 0; s >>= 1) {
        if (d < s) { for (int k = 0; k < 7; k++) red[k][d] += red[k][d + s]; }
        __syncthreads();
    }
    float iu  = 1.0f / fmaxf(sqrtf(red[3][0]), 1e-12f);
    float ium = 1.0f / fmaxf(sqrtf(red[4][0]), 1e-12f);
    float ip  = 1.0f / fmaxf(sqrtf(red[5][0]), 1e-12f);
    float ipm = 1.0f / fmaxf(sqrtf(red[6][0]), 1e-12f);
    float vu = u * iu, vm = umix * ium, vp = p * ip, vq = pmix * ipm;
    nueF[b * DIM + d] = vu;  nmuF[b * DIM + d] = vm;
    npeF[b * DIM + d] = vp;  nmpF[b * DIM + d] = vq;
    nueH[b * DIM + d] = (_Float16)vu;  nmuH[b * DIM + d] = (_Float16)vm;
    npeH[b * DIM + d] = (_Float16)vp;  nmpH[b * DIM + d] = (_Float16)vq;
    if (d == 0) {
        float ps = red[0][0], ns = red[1][0], nms = red[2][0];
        atomicAdd(&accums[2], neg_log_sigmoid(ps - ns));
        atomicAdd(&accums[3], neg_log_sigmoid(ps - nms));
        atomicAdd(&accums[0], bi);
    }
}

// coeff @ ue / coeff @ pe  -> single 64-vector each, normalized
__global__ __launch_bounds__(64) void k_cm(const float* __restrict__ ue,
        const float* __restrict__ pe, const float* __restrict__ cu,
        const float* __restrict__ cp, float* __restrict__ ncm) {
    int d = threadIdx.x;
    float su = 0.0f, sp = 0.0f;
    for (int b = 0; b < BATCH; b++) {
        su += cu[b] * ue[b * DIM + d];
        sp += cp[b] * pe[b * DIM + d];
    }
    __shared__ float r0[64], r1[64];
    r0[d] = su * su; r1[d] = sp * sp;
    __syncthreads();
    for (int s = 32; s > 0; s >>= 1) {
        if (d < s) { r0[d] += r0[d + s]; r1[d] += r1[d + s]; }
        __syncthreads();
    }
    ncm[d]      = su * (1.0f / fmaxf(sqrtf(r0[0]), 1e-12f));
    ncm[64 + d] = sp * (1.0f / fmaxf(sqrtf(r1[0]), 1e-12f));
}

// per-row diagonal / extra similarities needed by the NCE terms
__global__ __launch_bounds__(64) void k_sim(const float* __restrict__ nueF,
        const float* __restrict__ nmuF, const float* __restrict__ npeF,
        const float* __restrict__ nmpF, const float* __restrict__ ncm,
        const int* __restrict__ perm, float* __restrict__ sims) {
    int b = blockIdx.x, d = threadIdx.x;
    int pb = perm[b];
    float a  = nueF[b * DIM + d],  ap = nueF[pb * DIM + d], mu = nmuF[b * DIM + d];
    float e  = npeF[b * DIM + d],  ep = npeF[pb * DIM + d], mp = nmpF[b * DIM + d];
    __shared__ float red[8][64];
    red[0][d] = a * mu;          // psimU  = S_u[i,i]
    red[1][d] = ap * mu;         // psimNU = S_u[perm[i], i]
    red[2][d] = a * ap;          // dsimU
    red[3][d] = a * ncm[d];      // csimU
    red[4][d] = e * mp;          // psimP
    red[5][d] = ep * mp;         // psimNP
    red[6][d] = e * ep;          // dsimP
    red[7][d] = e * ncm[64 + d]; // csimP
    __syncthreads();
    for (int s = 32; s > 0; s >>= 1) {
        if (d < s) { for (int k = 0; k < 8; k++) red[k][d] += red[k][d + s]; }
        __syncthreads();
    }
    if (d < 8) sims[d * BATCH + b] = red[d][0];
}

// S = A @ B^T, A,B: [4096,64] f16 row-major, f32 accumulate.
// Block = 8 waves. A 16x64 panel staged in LDS via GLOBAL_LOAD_ASYNC_TO_LDS_B128
// (ASYNCcnt fence before barrier); each wave keeps its two A frags (full K=64)
// in VGPRs and sweeps 4 consecutive 16x16 n-tiles -> 8 v_wmma per wave.
__global__ __launch_bounds__(256) void k_gemm_nt(const _Float16* __restrict__ A,
        const _Float16* __restrict__ B, float* __restrict__ S) {
    __shared__ __align__(16) _Float16 aPanel[16 * 64];
    int tid = threadIdx.x;
    int mTile = blockIdx.y;
    if (tid < 128) {
        int r = tid >> 3, c = (tid & 7) * 8;
        async_b128_to_lds(&A[((size_t)mTile * 16 + r) * DIM + c], &aPanel[r * 64 + c]);
    }
    wait_asynccnt0();
    __syncthreads();

    int wave = tid >> 5, lane = tid & 31;
    int hsel = lane >> 4;        // half-wave select
    int r16  = lane & 15;
    const _Float16* aRow = &aPanel[r16 * 64];
    // A frag VGPR layout (16-bit A 16x32): lane<16 -> K {k0..k0+7},{k0+16..k0+23}
    union { v16h v; f32x4 f[2]; } a0, a1;
    a0.f[0] = *(const f32x4*)&aRow[ 0 + hsel * 8];
    a0.f[1] = *(const f32x4*)&aRow[16 + hsel * 8];
    a1.f[0] = *(const f32x4*)&aRow[32 + hsel * 8];
    a1.f[1] = *(const f32x4*)&aRow[48 + hsel * 8];

    int nBase = (blockIdx.x * 8 + wave) * 4;
    #pragma unroll
    for (int t = 0; t < 4; ++t) {
        int nTile = nBase + t;
        const _Float16* bRow = &B[((size_t)nTile * 16 + r16) * DIM];
        // B frag layout (16-bit B 32x16): lane<16 -> K {k0..k0+7},{k0+16..k0+23}; lane>=16 -> +8
        union { v16h v; f32x4 f[2]; } b0, b1;
        b0.f[0] = *(const f32x4*)&bRow[ 0 + hsel * 8];
        b0.f[1] = *(const f32x4*)&bRow[16 + hsel * 8];
        b1.f[0] = *(const f32x4*)&bRow[32 + hsel * 8];
        b1.f[1] = *(const f32x4*)&bRow[48 + hsel * 8];
        v8f c = {};
        c = wmma_f16(a0.v, b0.v, c);
        c = wmma_f16(a1.v, b1.v, c);
        int row0 = mTile * 16 + 8 * hsel;
        int col  = nTile * 16 + r16;
        #pragma unroll
        for (int r = 0; r < 8; ++r)
            S[(size_t)(row0 + r) * BATCH + col] = c[r];
    }
}

// row-wise max + sum(exp) of S/TEMP (one block per row)
__global__ __launch_bounds__(256) void k_lse(const float* __restrict__ S,
        float* __restrict__ rowM, float* __restrict__ rowE) {
    int row = blockIdx.x, t = threadIdx.x;
    const float* Sr = S + (size_t)row * BATCH;
    __shared__ float sm[256];
    float m = -1e30f;
    for (int j = t; j < BATCH; j += 256) m = fmaxf(m, Sr[j] * INV_TEMP);
    sm[t] = m; __syncthreads();
    for (int s = 128; s > 0; s >>= 1) { if (t < s) sm[t] = fmaxf(sm[t], sm[t + s]); __syncthreads(); }
    float M = sm[0]; __syncthreads();
    float e = 0.0f;
    for (int j = t; j < BATCH; j += 256) e += expf(Sr[j] * INV_TEMP - M);
    sm[t] = e; __syncthreads();
    for (int s = 128; s > 0; s >>= 1) { if (t < s) sm[t] += sm[t + s]; __syncthreads(); }
    if (t == 0) { rowM[row] = M; rowE[row] = sm[0]; }
}

// dual contrastive loss accumulation; l_neg reuses the l_pos LSE stats at perm[i]
__global__ __launch_bounds__(256) void k_cl(const float* __restrict__ sims,
        const float* __restrict__ rowMU, const float* __restrict__ rowEU,
        const float* __restrict__ rowMP, const float* __restrict__ rowEP,
        const int* __restrict__ perm, const float* __restrict__ beta_u,
        const float* __restrict__ beta_i, float* __restrict__ accums) {
    int i = blockIdx.x * blockDim.x + threadIdx.x;
    if (i >= BATCH) return;
    int j = perm[i];
    const float* psimU  = sims;              const float* psimNU = sims + BATCH;
    const float* dsimU  = sims + 2 * BATCH;  const float* csimU  = sims + 3 * BATCH;
    const float* psimP  = sims + 4 * BATCH;  const float* psimNP = sims + 5 * BATCH;
    const float* dsimP  = sims + 6 * BATCH;  const float* csimP  = sims + 7 * BATCH;

    float lposU = -psimU[i] * INV_TEMP +
        lse3(rowMU[i], rowEU[i], dsimU[i] * INV_TEMP, csimU[i] * INV_TEMP);
    float lnegU = -psimNU[i] * INV_TEMP +
        lse3(rowMU[j], rowEU[j], dsimU[i] * INV_TEMP, csimU[j] * INV_TEMP);
    atomicAdd(&accums[4], beta_u[i] * lposU + (1.0f - beta_u[i]) * lnegU);

    float lposP = -psimP[i] * INV_TEMP +
        lse3(rowMP[i], rowEP[i], dsimP[i] * INV_TEMP, csimP[i] * INV_TEMP);
    float lnegP = -psimNP[i] * INV_TEMP +
        lse3(rowMP[j], rowEP[j], dsimP[i] * INV_TEMP, csimP[j] * INV_TEMP);
    atomicAdd(&accums[5], beta_i[i] * lposP + (1.0f - beta_i[i]) * lnegP);
}

__global__ void k_final(const float* __restrict__ accums, float* __restrict__ out) {
    float bbar = accums[0] * (1.0f / BATCH);
    float main_ = bbar * accums[2] * (1.0f / BATCH) + (1.0f - bbar) * accums[3] * (1.0f / BATCH);
    float cl   = SSL_LAMBDA * (accums[4] + accums[5]) * (1.0f / BATCH);
    float reg  = REG_C * accums[1] * (1.0f / BATCH);
    out[0] = main_ + cl + reg;
}

// ---------------- host launch ----------------
extern "C" void kernel_launch(void* const* d_in, const int* in_sizes, int n_in,
                              void* d_out, int out_size, void* d_ws, size_t ws_size,
                              hipStream_t stream) {
    (void)n_in; (void)out_size; (void)ws_size;
    const float* user_emb = (const float*)d_in[0];
    const float* item_emb = (const float*)d_in[1];
    const int*   adj_rows = (const int*)d_in[2];
    const int*   adj_cols = (const int*)d_in[3];
    const float* adj_vals = (const float*)d_in[4];
    const int*   user     = (const int*)d_in[5];
    const int*   pos      = (const int*)d_in[6];
    const int*   neg      = (const int*)d_in[7];
    const float* beta_u   = (const float*)d_in[8];
    const float* beta_i   = (const float*)d_in[9];
    const int*   perm     = (const int*)d_in[10];
    const float* coeff_u  = (const float*)d_in[11];
    const float* coeff_p  = (const float*)d_in[12];
    int nnz = in_sizes[2];

    float* ws   = (float*)d_ws;
    float* acc  = ws + OFF_ACC;
    float* cur  = ws + OFF_CUR;
    float* nxt  = ws + OFF_NXT;
    float* S    = ws + OFF_S;
    float* ue   = ws + OFF_UE,   *pe = ws + OFF_PE,   *ne = ws + OFF_NE;
    float* nueF = ws + OFF_NUEF, *nmuF = ws + OFF_NMUF;
    float* npeF = ws + OFF_NPEF, *nmpF = ws + OFF_NMPF;
    _Float16* h16  = (_Float16*)(ws + OFF_H16);
    _Float16* nueH = h16;             _Float16* nmuH = h16 + BROW;
    _Float16* npeH = h16 + 2 * BROW;  _Float16* nmpH = h16 + 3 * BROW;
    float* ncm   = ws + OFF_NCM;
    float* sims  = ws + OFF_SIMS;
    float* rowMU = ws + OFF_ROWS;            float* rowEU = rowMU + BATCH;
    float* rowMP = rowMU + 2 * BATCH;        float* rowEP = rowMU + 3 * BATCH;
    float* accums = ws + OFF_ACCS;

    const int T = 256;
    int nodeBlocks = (int)((NN64 + T - 1) / T);
    int edgeBlocks = (int)(((size_t)nnz * 32 + T - 1) / T);

    k_zero<<<1, 64, 0, stream>>>(accums, 64);
    k_init<<<nodeBlocks, T, 0, stream>>>(user_emb, item_emb, acc, cur);
    for (int l = 0; l < N_LAYERS; ++l) {
        k_zero<<<nodeBlocks, T, 0, stream>>>(nxt, NN64);
        k_scatter<<<edgeBlocks, T, 0, stream>>>(adj_rows, adj_cols, adj_vals, cur, nxt, nnz);
        k_update<<<nodeBlocks, T, 0, stream>>>(acc, cur, nxt);
    }
    k_gather<<<BATCH, 64, 0, stream>>>(acc, user, pos, neg, user_emb, item_emb,
                                       ue, pe, ne, accums);
    k_batch<<<BATCH, 64, 0, stream>>>(ue, pe, ne, perm, beta_u, beta_i,
                                      nueF, nmuF, npeF, nmpF,
                                      nueH, nmuH, npeH, nmpH, accums);
    k_cm<<<1, 64, 0, stream>>>(ue, pe, coeff_u, coeff_p, ncm);
    k_sim<<<BATCH, 64, 0, stream>>>(nueF, nmuF, npeF, nmpF, ncm, perm, sims);

    // 8 waves/block, 4 n-tiles/wave -> grid (4096/16/8/4, 4096/16) = (8, 256)
    dim3 gGrid(BATCH / 16 / 8 / 4, BATCH / 16);
    k_gemm_nt<<<gGrid, 256, 0, stream>>>(nueH, nmuH, S);
    k_lse<<<BATCH, 256, 0, stream>>>(S, rowMU, rowEU);
    k_gemm_nt<<<gGrid, 256, 0, stream>>>(npeH, nmpH, S);
    k_lse<<<BATCH, 256, 0, stream>>>(S, rowMP, rowEP);

    k_cl<<<(BATCH + T - 1) / T, T, 0, stream>>>(sims, rowMU, rowEU, rowMP, rowEP,
                                                perm, beta_u, beta_i, accums);
    k_final<<<1, 1, 0, stream>>>(accums, (float*)d_out);
}
